// HoneycombConv2d_42133629174434
// MI455X (gfx1250) — matrix operand order repaired and checked by reading, not compiled, not run on hardware
//
#include <hip/hip_runtime.h>
#include <hip/hip_bf16.h>

// HoneycombConv2d as implicit GEMM on CDNA5 WMMA bf16 (fp32 split hi/lo, 3 products).
// B=8, CI=CO=128, H=W=128, K=7 hex taps. d_ws requirement: 458,752 bytes.

typedef __attribute__((ext_vector_type(16))) __bf16 v16bf;
typedef __attribute__((ext_vector_type(8)))  float  v8f;

#define B_  8
#define C_  128
#define H_  128
#define W_  128
#define NW  114688            // 7*128*128 weight elements

// LDS layout (bytes)
#define XSTR   272            // per (slice,w) row: 128 bf16 (256B) + 16B pad
#define XSLICE (128 * XSTR)   // 34816
#define XHI    0
#define XLO    (3 * XSLICE)          // 104448
#define WHI    (6 * XSLICE)          // 208896
#define WLO    (WHI + 128 * XSTR)    // 243712
#define LDS_BYTES (WLO + 128 * XSTR) // 278528  (< 320KB per WGP)
#define OROW   528            // epilogue stage row: 128 f32 + 16B pad (reuses x region)

static __device__ __forceinline__ unsigned short bf16_rne(float f) {
  unsigned int u = __float_as_uint(f);
  u += 0x7FFFu + ((u >> 16) & 1u);
  return (unsigned short)(u >> 16);
}
static __device__ __forceinline__ float bf16f(unsigned short h) {
  return __uint_as_float(((unsigned int)h) << 16);
}

union Frag { v16bf v; uint4 u[2]; };

// ---- prep: weight[o][i][t] fp32 -> whi/wlo [t][o][i] bf16 (transposed, o-major) ----
__global__ void prep_w(const float* __restrict__ w,
                       unsigned short* __restrict__ whi,
                       unsigned short* __restrict__ wlo) {
  int e = blockIdx.x * 256 + threadIdx.x;
  if (e >= NW) return;
  int t = e / 16384;
  int o = (e >> 7) & 127;
  int i = e & 127;
  float f = w[(o * 128 + i) * 7 + t];
  unsigned short hi = bf16_rne(f);
  whi[e] = hi;
  wlo[e] = bf16_rne(f - bf16f(hi));
}

// ---- main: one block per (b, h); 8 waves, wave n owns o-tile n; M = full w row ----
__global__ void __launch_bounds__(256)
honeycomb_wmma(const float* __restrict__ x,
               const unsigned short* __restrict__ gwhi,
               const unsigned short* __restrict__ gwlo,
               const float* __restrict__ bias,
               float* __restrict__ out) {
  extern __shared__ char smem[];
  const int tid  = threadIdx.x;
  const int b    = blockIdx.x >> 7;
  const int h    = blockIdx.x & 127;
  const int lane = tid & 31;
  const int wv   = tid >> 5;     // wave id 0..7
  const int lh   = lane & 15;
  const int hs   = lane >> 4;    // half-wave select
  const int o0   = wv * 16;

  // ---- stage x rows h+1, h, h-1 (slice s = dq+1) as hi/lo bf16, layout [s][w][i] ----
  for (int e = tid; e < 3 * C_ * W_; e += 256) {
    int s = e >> 14;             // 16384 elements per slice
    int r = e & 16383;
    int i = r >> 7;
    int w = r & 127;
    int hh = (h + 1 - s) & 127;  // row h - dq, dq = s-1
    float f = x[((b * C_ + i) * H_ + hh) * W_ + w];
    unsigned short hi = bf16_rne(f);
    unsigned short lo = bf16_rne(f - bf16f(hi));
    int off = s * XSLICE + w * XSTR + i * 2;
    *(unsigned short*)(smem + XHI + off) = hi;
    *(unsigned short*)(smem + XLO + off) = lo;
  }

  v8f acc[8];
  #pragma unroll
  for (int m = 0; m < 8; ++m) acc[m] = v8f{};

  // taps in reference order: (dq,dr) = (-1,0)(-1,1)(0,-1)(0,0)(0,1)(1,-1)(1,0)
  // s = dq+1 nibbles (t6..t0):   2 2 1 1 1 0 0  -> 0x2211100
  // dr+1 nibbles (t6..t0):       1 0 2 1 0 2 1  -> 0x1021021
  for (int t = 0; t < 7; ++t) {
    __syncthreads();             // previous tap's weight reads done
    // stage tap-t weights: straight 16B copies into padded [o][i] rows
    for (int c = tid; c < 2048; c += 256) {
      int o  = c >> 4;
      int i0 = (c & 15) * 8;
      int g  = t * 16384 + o * 128 + i0;
      int l  = o * XSTR + i0 * 2;
      *(uint4*)(smem + WHI + l) = *(const uint4*)(gwhi + g);
      *(uint4*)(smem + WLO + l) = *(const uint4*)(gwlo + g);
    }
    if (t < 6) {                 // pull next tap's weights toward L2/L0
      __builtin_prefetch(gwhi + (t + 1) * 16384 + tid * 64, 0, 1);
      __builtin_prefetch(gwlo + (t + 1) * 16384 + tid * 64, 0, 1);
    }
    __syncthreads();

    const int s  = (0x2211100 >> (4 * t)) & 0xF;
    const int dr = ((0x1021021 >> (4 * t)) & 0xF) - 1;

    for (int c4 = 0; c4 < 4; ++c4) {       // K chunks of 32 channels
      const int i0 = c4 * 32;
      // B fragment: lanes<16 hold K=0..15 (i0..i0+15), lanes>=16 K=16..31
      Frag bh, bl;
      {
        int boff = (o0 + lh) * XSTR + (i0 + hs * 16) * 2;
        bh.u[0] = *(const uint4*)(smem + WHI + boff);
        bh.u[1] = *(const uint4*)(smem + WHI + boff + 16);
        bl.u[0] = *(const uint4*)(smem + WLO + boff);
        bl.u[1] = *(const uint4*)(smem + WLO + boff + 16);
      }
      #pragma unroll
      for (int m = 0; m < 8; ++m) {        // 8 M-tiles = whole w row
        // A fragment: M = lane&15 (w position); lanes<16 K 0-15, lanes>=16 K 16-31
        int wrow = (m * 16 + lh - dr + 128) & 127;   // circular w - dr
        int aoff = s * XSLICE + wrow * XSTR + (i0 + hs * 16) * 2;
        Frag ah, al;
        ah.u[0] = *(const uint4*)(smem + XHI + aoff);
        ah.u[1] = *(const uint4*)(smem + XHI + aoff + 16);
        al.u[0] = *(const uint4*)(smem + XLO + aoff);
        al.u[1] = *(const uint4*)(smem + XLO + aoff + 16);
        acc[m] = __builtin_amdgcn_wmma_f32_16x16x32_bf16(
                     false, ah.v, false, bh.v, (short)0, acc[m], false, false);
        acc[m] = __builtin_amdgcn_wmma_f32_16x16x32_bf16(
                     false, ah.v, false, bl.v, (short)0, acc[m], false, false);
        acc[m] = __builtin_amdgcn_wmma_f32_16x16x32_bf16(
                     false, al.v, false, bh.v, (short)0, acc[m], false, false);
      }
    }
  }

  // ---- epilogue: D layout (vgpr r, lane) = (w = m*16+r+hs*8, o = o0+lh);
  //      transpose through LDS so global stores are coalesced 512B rows ----
  __syncthreads();
  #pragma unroll
  for (int m = 0; m < 8; ++m) {
    #pragma unroll
    for (int r = 0; r < 8; ++r) {
      int o = o0 + lh;
      int w = m * 16 + r + hs * 8;
      *(float*)(smem + o * OROW + w * 4) = acc[m][r];
    }
  }
  __syncthreads();
  for (int e = tid; e < C_ * W_; e += 256) {
    int o = e >> 7;
    int w = e & 127;
    float v = *(const float*)(smem + o * OROW + w * 4) + bias[o];
    out[((b * C_ + o) * H_ + h) * W_ + w] = v;
  }
}

extern "C" void kernel_launch(void* const* d_in, const int* in_sizes, int n_in,
                              void* d_out, int out_size, void* d_ws, size_t ws_size,
                              hipStream_t stream) {
  const float* x    = (const float*)d_in[0];
  const float* wt   = (const float*)d_in[1];
  const float* bias = (const float*)d_in[2];
  float* out = (float*)d_out;

  unsigned short* whi = (unsigned short*)d_ws;      // [7][128][128] bf16 hi
  unsigned short* wlo = whi + NW;                   // [7][128][128] bf16 lo

  prep_w<<<(NW + 255) / 256, 256, 0, stream>>>(wt, whi, wlo);
  honeycomb_wmma<<<dim3(B_ * H_), dim3(256), LDS_BYTES, stream>>>(x, whi, wlo, bias, out);
}
